// MetaFiber_38431367365147
// MI455X (gfx1250) — compile-verified, matrix operand order checked by reading, not run on hardware
//
#include <hip/hip_runtime.h>
#include <hip/hip_bf16.h>

#define S_STEPS 20
#define NFFT    4096
#define BATCH   512
#define WID     60
#define WIDP    64
#define SPS     8
#define PHI     0.3f
#define SLOPE   0.01f
#define PI_F    3.14159265358979323846f

typedef __attribute__((ext_vector_type(16))) _Float16 v16h;
typedef __attribute__((ext_vector_type(8)))  float    v8f;

// ---------------------------------------------------------------------------
// Pack a lane-local array of 16 scalars (two contiguous K-runs of 8) into the
// 16-bit A/B WMMA fragment layout:
//   VGPR r (halfs 2r,2r+1) holds K = (r<4?0:16) + g*8 + 2*(r&3), +1
//   relative to the runs: r<4 -> run0[2*(r&3)..], r>=4 -> run1[2*(r&3)..]
// ---------------------------------------------------------------------------
__device__ inline v16h pack_frag(const float* v) {
    v16h f;
#pragma unroll
    for (int r = 0; r < 8; ++r) {
        int j0 = ((r & 4) ? 8 : 0) + 2 * (r & 3);
        f[2 * r]     = (_Float16)v[j0];
        f[2 * r + 1] = (_Float16)v[j0 + 1];
    }
    return f;
}

// ---------------------------------------------------------------------------
// Zero-padded repack:  dst[r, c] (rows_pad x cols_pad, row-major)
//                    = src[r, c] if r < rows_act && c < cols_act else 0
// ---------------------------------------------------------------------------
__global__ __launch_bounds__(256) void pad2d_kernel(
    const float2* __restrict__ src, float2* __restrict__ dst,
    int rows_act, int cols_act, int src_ld, int rows_pad, int cols_pad)
{
    const int idx = blockIdx.x * blockDim.x + threadIdx.x;
    if (idx >= rows_pad * cols_pad) return;
    const int r = idx / cols_pad, c = idx % cols_pad;
    float2 v = make_float2(0.f, 0.f);
    if (r < rows_act && c < cols_act) v = src[(size_t)r * src_ld + c];
    dst[idx] = v;
}

// ---------------------------------------------------------------------------
// Complex GEMM on padded operands:  Y[m,n] = sum_k A[m,k] * W[n,k] (+bias)
//   AMODE 0: A complex row-major, row stride lda (complex elems)
//   AMODE 1: A[m,k] = 0.5*(u[m,k,0]+u[m,k,1]) read as one float4 (pol mean)
// W is pre-padded: exactly (ntiles*16) rows x K cols, K % 32 == 0.
// Branch-free inner loop, float4 (b128) operand loads, 4 WMMAs / K-chunk.
// ---------------------------------------------------------------------------
template <int AMODE>
__global__ __launch_bounds__(256) void cgemm_wmma_kernel(
    const float* __restrict__ Abase, int lda,
    const float2* __restrict__ W, int ldw,
    const float2* __restrict__ bias,
    float2* __restrict__ Y, int ldo,
    int ntiles, int K, int relu)
{
    const int lane = threadIdx.x & 31;
    const int wid  = blockIdx.x * (blockDim.x >> 5) + (threadIdx.x >> 5);
    const int mt = wid / ntiles, nt = wid % ntiles;
    const int g = lane >> 4, hh = lane & 15;
    const int mrow = mt * 16 + hh;        // A row owned by this lane
    const int ncol = nt * 16 + hh;        // W row (= output column)

    const float2* Arow = (const float2*)Abase + (size_t)mrow * lda;  // AMODE 0
    const float4* Urow = (const float4*)Abase + (size_t)mrow * NFFT; // AMODE 1
    const float2* Wrow = W + (size_t)ncol * ldw;

    v8f accP = {}, accQ = {}, accI = {};

    for (int k0 = 0; k0 < K; k0 += 32) {
        if (k0 + 32 < K)
            __builtin_prefetch((const void*)(Wrow + k0 + 32), 0, 1); // global_prefetch_b8

        float ar[16], ai[16], br[16], bi[16];
#pragma unroll
        for (int q = 0; q < 2; ++q) {
            const int kb = k0 + q * 16 + g * 8;   // 8-complex run, 16B aligned
            if (AMODE == 1) {
#pragma unroll
                for (int j = 0; j < 8; ++j) {
                    float4 v = Urow[kb + j];      // (u0r,u0i,u1r,u1i)
                    ar[8 * q + j] = 0.5f * (v.x + v.z);
                    ai[8 * q + j] = 0.5f * (v.y + v.w);
                }
            } else {
                const float4* p = (const float4*)(Arow + kb);
#pragma unroll
                for (int j = 0; j < 4; ++j) {
                    float4 v = p[j];              // two complex per b128
                    ar[8 * q + 2 * j]     = v.x;  ai[8 * q + 2 * j]     = v.y;
                    ar[8 * q + 2 * j + 1] = v.z;  ai[8 * q + 2 * j + 1] = v.w;
                }
            }
            const float4* pw = (const float4*)(Wrow + kb);
#pragma unroll
            for (int j = 0; j < 4; ++j) {
                float4 v = pw[j];
                br[8 * q + 2 * j]     = v.x;  bi[8 * q + 2 * j]     = v.y;
                br[8 * q + 2 * j + 1] = v.z;  bi[8 * q + 2 * j + 1] = v.w;
            }
        }
        v16h aR = pack_frag(ar), aI = pack_frag(ai);
        v16h bR = pack_frag(br), bI = pack_frag(bi);
        // Yr = Ar*Br - Ai*Bi ; Yi = Ar*Bi + Ai*Br
        accP = __builtin_amdgcn_wmma_f32_16x16x32_f16(false, aR, false, bR, (short)0, accP, false, false);
        accQ = __builtin_amdgcn_wmma_f32_16x16x32_f16(false, aI, false, bI, (short)0, accQ, false, false);
        accI = __builtin_amdgcn_wmma_f32_16x16x32_f16(false, aR, false, bI, (short)0, accI, false, false);
        accI = __builtin_amdgcn_wmma_f32_16x16x32_f16(false, aI, false, bR, (short)0, accI, false, false);
    }

    const int n = nt * 16 + hh;           // C/D layout: col = lane&15
    const float2 bb = bias[n];            // bias pre-padded
#pragma unroll
    for (int j = 0; j < 8; ++j) {         // C/D layout: row = j + 8*g
        const int m = mt * 16 + j + 8 * g;
        float yr = accP[j] - accQ[j] + bb.x;
        float yi = accI[j] + bb.y;
        if (relu) {
            yr = yr > 0.f ? yr : SLOPE * yr;
            yi = yi > 0.f ? yi : SLOPE * yi;
        }
        Y[(size_t)m * ldo + n] = make_float2(yr, yi);
    }
}

// ---------------------------------------------------------------------------
// 4096-pt Stockham radix-2 FFT in LDS (auto-sort, no bit reversal).
// 12 stages, 512 threads * 4 butterflies/stage. Returns buffer holding result.
// ---------------------------------------------------------------------------
__device__ inline float2* fft12(float2* x, float2* y, int tid) {
#pragma unroll 1
    for (int st = 0; st < 12; ++st) {
        const int m = 1 << st;
        for (int t = tid; t < 2048; t += 512) {
            const int jm = t & ~(m - 1);          // j*m
            const int k  = t & (m - 1);
            float2 c0 = x[t];
            float2 c1 = x[t + 2048];
            float sn, cs;
            __sincosf(-PI_F * (1.0f / 2048.0f) * (float)jm, &sn, &cs);
            float2 dsum = make_float2(c0.x + c1.x, c0.y + c1.y);
            float2 ddif = make_float2(c0.x - c1.x, c0.y - c1.y);
            const int od = k + (jm << 1);
            y[od]     = dsum;
            y[od + m] = make_float2(ddif.x * cs - ddif.y * sn,
                                    ddif.x * sn + ddif.y * cs);
        }
        __syncthreads();
        float2* tmp = x; x = y; y = tmp;
    }
    return x;
}

// Linear SSFM step: u <- IFFT( FFT(u) * (filtFNN + H) ), both polarizations.
// blockIdx.x = b*2 + pol. Inverse via conjugation trick.
// Input staged into LDS with GLOBAL_LOAD_ASYNC_TO_LDS_B64 (per-lane addresses
// handle the pol-interleaved stride), drained via s_wait_asynccnt.
__global__ __launch_bounds__(512) void fft_filter_kernel(
    float2* __restrict__ u, const float2* __restrict__ filt,
    const float2* __restrict__ Hd)
{
    __shared__ float2 bufA[NFFT];
    __shared__ float2 bufB[NFFT];
    const int tid = threadIdx.x;
    const int b = blockIdx.x >> 1, pol = blockIdx.x & 1;
    float2* base = u + (size_t)b * NFFT * 2 + pol;

    // ---- async global -> LDS staging (ASYNCcnt-tracked DMA path) ----
#pragma unroll
    for (int i = tid; i < NFFT; i += 512) {
        unsigned lds_addr = (unsigned)(unsigned long long)&bufA[i];
        unsigned long long gaddr = (unsigned long long)(base + (size_t)i * 2);
        asm volatile("global_load_async_to_lds_b64 %0, %1, off"
                     :: "v"(lds_addr), "v"(gaddr) : "memory");
    }
    asm volatile("s_wait_asynccnt 0" ::: "memory");
    __syncthreads();

    float2* x = fft12(bufA, bufB, tid);
    float2* y = (x == bufA) ? bufB : bufA;

    const float2* fr = filt + (size_t)b * NFFT;
    for (int i = tid; i < NFFT; i += 512) {
        float2 F = fr[i], Hh = Hd[i];
        F.x += Hh.x; F.y += Hh.y;
        float2 v = x[i];
        float2 p = make_float2(v.x * F.x - v.y * F.y,
                               v.x * F.y + v.y * F.x);
        x[i] = make_float2(p.x, -p.y);          // conj -> inverse via fwd FFT
    }
    __syncthreads();

    x = fft12(x, y, tid);
    const float sc = 1.0f / (float)NFFT;
    for (int i = tid; i < NFFT; i += 512) {
        float2 v = x[i];
        base[(size_t)i * 2] = make_float2(v.x * sc, -v.y * sc);
    }
}

// sc[b] = |h[b,:60] . Sw1 + Sb1|^2 + 1
__global__ __launch_bounds__(256) void sc_kernel(
    const float2* __restrict__ h, const float2* __restrict__ w1,
    const float2* __restrict__ b1, float* __restrict__ sc)
{
    const int b = blockIdx.x * blockDim.x + threadIdx.x;
    if (b >= BATCH) return;
    float2 bb = *b1;
    float yr = bb.x, yi = bb.y;
#pragma unroll 4
    for (int k = 0; k < WID; ++k) {
        float2 a = h[(size_t)b * WIDP + k];
        float2 w = w1[k];
        yr += a.x * w.x - a.y * w.y;
        yi += a.x * w.y + a.y * w.x;
    }
    sc[b] = yr * yr + yi * yi + 1.0f;
}

// u *= exp(-i * sc[b] * PHI * (|u0|^2+|u1|^2)) per (b,n), both polarizations.
__global__ __launch_bounds__(256) void phase_kernel(
    float2* __restrict__ u, const float* __restrict__ sc)
{
    const int idx = blockIdx.x * blockDim.x + threadIdx.x;   // b*NFFT + n
    const int b = idx >> 12;
    float4 v = ((float4*)u)[idx];
    const float power = v.x * v.x + v.y * v.y + v.z * v.z + v.w * v.w;
    float sn, cs;
    __sincosf(-sc[b] * PHI * power, &sn, &cs);
    float4 o;
    o.x = v.x * cs - v.y * sn;  o.y = v.x * sn + v.y * cs;
    o.z = v.z * cs - v.w * sn;  o.w = v.z * sn + v.w * cs;
    ((float4*)u)[idx] = o;
}

// Complex conv1d, kernel=stride=8, out [B, 512, 2] complex.
__global__ __launch_bounds__(256) void conv_kernel(
    const float2* __restrict__ u,
    const float* __restrict__ Cwr, const float* __restrict__ Cwi,
    const float* __restrict__ Cbr, const float* __restrict__ Cbi,
    float2* __restrict__ out)
{
    const int t = blockIdx.x * blockDim.x + threadIdx.x;     // b*512 + tt
    const int b = t >> 9, tt = t & 511;
#pragma unroll
    for (int oc = 0; oc < 2; ++oc) {
        float rr = Cbr[oc] - Cbi[oc];
        float ri = Cbr[oc] + Cbi[oc];
#pragma unroll
        for (int ic = 0; ic < 2; ++ic) {
#pragma unroll
            for (int k = 0; k < SPS; ++k) {
                float2 xv = u[((size_t)b * NFFT + tt * SPS + k) * 2 + ic];
                float wr = Cwr[(oc * 2 + ic) * SPS + k];
                float wi = Cwi[(oc * 2 + ic) * SPS + k];
                rr += xv.x * wr - xv.y * wi;
                ri += xv.y * wr + xv.x * wi;
            }
        }
        out[((size_t)b * 512 + tt) * 2 + oc] = make_float2(rr, ri);
    }
}

// ---------------------------------------------------------------------------
extern "C" void kernel_launch(void* const* d_in, const int* in_sizes, int n_in,
                              void* d_out, int out_size, void* d_ws, size_t ws_size,
                              hipStream_t stream) {
    (void)in_sizes; (void)n_in; (void)out_size; (void)ws_size;
    const float*  u_in = (const float*) d_in[0];
    const float2* H    = (const float2*)d_in[1];
    const float2* Hw0  = (const float2*)d_in[2];
    const float2* Hb0  = (const float2*)d_in[3];
    const float2* Hwh  = (const float2*)d_in[4];
    const float2* Hbh  = (const float2*)d_in[5];
    const float2* Hw1  = (const float2*)d_in[6];
    const float2* Hb1  = (const float2*)d_in[7];
    const float2* Sw0  = (const float2*)d_in[8];
    const float2* Sb0  = (const float2*)d_in[9];
    const float2* Swh  = (const float2*)d_in[10];
    const float2* Sbh  = (const float2*)d_in[11];
    const float2* Sw1  = (const float2*)d_in[12];
    const float2* Sb1  = (const float2*)d_in[13];
    const float*  Cwr  = (const float*) d_in[14];
    const float*  Cwi  = (const float*) d_in[15];
    const float*  Cbr  = (const float*) d_in[16];
    const float*  Cbi  = (const float*) d_in[17];

    // workspace layout
    char* ws = (char*)d_ws;
    float2* u     = (float2*)ws; ws += (size_t)BATCH * NFFT * 2 * sizeof(float2); // 32 MB
    float2* filt  = (float2*)ws; ws += (size_t)BATCH * NFFT * sizeof(float2);     // 16 MB
    float2* Hw0p  = (float2*)ws; ws += (size_t)WIDP * NFFT * sizeof(float2);      // 2 MB
    float2* Hw1p  = (float2*)ws; ws += (size_t)NFFT * WIDP * sizeof(float2);      // 2 MB
    float2* Sw0p  = (float2*)ws; ws += (size_t)WIDP * 2 * NFFT * sizeof(float2);  // 4 MB
    float2* Whp0  = (float2*)ws; ws += (size_t)WIDP * WIDP * sizeof(float2);
    float2* Whp1  = (float2*)ws; ws += (size_t)WIDP * WIDP * sizeof(float2);
    float2* Swp0  = (float2*)ws; ws += (size_t)WIDP * WIDP * sizeof(float2);
    float2* Swp1  = (float2*)ws; ws += (size_t)WIDP * WIDP * sizeof(float2);
    float2* bHb0  = (float2*)ws; ws += WIDP * sizeof(float2);
    float2* bHbh0 = (float2*)ws; ws += WIDP * sizeof(float2);
    float2* bHbh1 = (float2*)ws; ws += WIDP * sizeof(float2);
    float2* bSb0  = (float2*)ws; ws += WIDP * sizeof(float2);
    float2* bSbh0 = (float2*)ws; ws += WIDP * sizeof(float2);
    float2* bSbh1 = (float2*)ws; ws += WIDP * sizeof(float2);
    float2* hA    = (float2*)ws; ws += (size_t)BATCH * WIDP * sizeof(float2);
    float2* hB    = (float2*)ws; ws += (size_t)BATCH * WIDP * sizeof(float2);
    float*  scb   = (float*)ws;  ws += (size_t)BATCH * sizeof(float);

    hipMemcpyAsync(u, u_in, (size_t)BATCH * NFFT * 2 * sizeof(float2),
                   hipMemcpyDeviceToDevice, stream);

    for (int s = 0; s < S_STEPS; ++s) {
        // ---- repack this step's weights/biases into zero-padded layouts ----
        pad2d_kernel<<<(WIDP * NFFT) / 256, 256, 0, stream>>>(
            Hw0 + (size_t)s * WID * NFFT, Hw0p, WID, NFFT, NFFT, WIDP, NFFT);
        pad2d_kernel<<<(NFFT * WIDP) / 256, 256, 0, stream>>>(
            Hw1 + (size_t)s * NFFT * WID, Hw1p, NFFT, WID, WID, NFFT, WIDP);
        pad2d_kernel<<<(WIDP * 2 * NFFT) / 256, 256, 0, stream>>>(
            Sw0 + (size_t)s * WID * 2 * NFFT, Sw0p, WID, 2 * NFFT, 2 * NFFT, WIDP, 2 * NFFT);
        pad2d_kernel<<<(WIDP * WIDP) / 256, 256, 0, stream>>>(
            Hwh + ((size_t)s * 2 + 0) * WID * WID, Whp0, WID, WID, WID, WIDP, WIDP);
        pad2d_kernel<<<(WIDP * WIDP) / 256, 256, 0, stream>>>(
            Hwh + ((size_t)s * 2 + 1) * WID * WID, Whp1, WID, WID, WID, WIDP, WIDP);
        pad2d_kernel<<<(WIDP * WIDP) / 256, 256, 0, stream>>>(
            Swh + ((size_t)s * 2 + 0) * WID * WID, Swp0, WID, WID, WID, WIDP, WIDP);
        pad2d_kernel<<<(WIDP * WIDP) / 256, 256, 0, stream>>>(
            Swh + ((size_t)s * 2 + 1) * WID * WID, Swp1, WID, WID, WID, WIDP, WIDP);
        pad2d_kernel<<<1, 256, 0, stream>>>(Hb0 + (size_t)s * WID, bHb0, 1, WID, WID, 1, WIDP);
        pad2d_kernel<<<1, 256, 0, stream>>>(Hbh + ((size_t)s * 2 + 0) * WID, bHbh0, 1, WID, WID, 1, WIDP);
        pad2d_kernel<<<1, 256, 0, stream>>>(Hbh + ((size_t)s * 2 + 1) * WID, bHbh1, 1, WID, WID, 1, WIDP);
        pad2d_kernel<<<1, 256, 0, stream>>>(Sb0 + (size_t)s * WID, bSb0, 1, WID, WID, 1, WIDP);
        pad2d_kernel<<<1, 256, 0, stream>>>(Sbh + ((size_t)s * 2 + 0) * WID, bSbh0, 1, WID, WID, 1, WIDP);
        pad2d_kernel<<<1, 256, 0, stream>>>(Sbh + ((size_t)s * 2 + 1) * WID, bSbh1, 1, WID, WID, 1, WIDP);

        // ---- filter FNN: fc0 (pol-mean folded into A load) ----
        cgemm_wmma_kernel<1><<<16, 256, 0, stream>>>(
            (const float*)u, 0, Hw0p, NFFT, bHb0, hA, WIDP, 4, NFFT, 1);
        cgemm_wmma_kernel<0><<<16, 256, 0, stream>>>(
            (const float*)hA, WIDP, Whp0, WIDP, bHbh0, hB, WIDP, 4, WIDP, 1);
        cgemm_wmma_kernel<0><<<16, 256, 0, stream>>>(
            (const float*)hB, WIDP, Whp1, WIDP, bHbh1, hA, WIDP, 4, WIDP, 1);
        cgemm_wmma_kernel<0><<<1024, 256, 0, stream>>>(
            (const float*)hA, WIDP, Hw1p, WIDP, Hb1 + (size_t)s * NFFT,
            filt, NFFT, 256, WIDP, 0);
        // ---- linear step: FFT * (filt + H) * IFFT ----
        fft_filter_kernel<<<BATCH * 2, 512, 0, stream>>>(u, filt, H);
        // ---- scalar FNN on flat u [B, 2*NFFT] ----
        cgemm_wmma_kernel<0><<<16, 256, 0, stream>>>(
            (const float*)u, 2 * NFFT, Sw0p, 2 * NFFT, bSb0, hA, WIDP, 4, 2 * NFFT, 1);
        cgemm_wmma_kernel<0><<<16, 256, 0, stream>>>(
            (const float*)hA, WIDP, Swp0, WIDP, bSbh0, hB, WIDP, 4, WIDP, 1);
        cgemm_wmma_kernel<0><<<16, 256, 0, stream>>>(
            (const float*)hB, WIDP, Swp1, WIDP, bSbh1, hA, WIDP, 4, WIDP, 1);
        sc_kernel<<<2, 256, 0, stream>>>(hA, Sw1 + (size_t)s * WID, Sb1 + s, scb);
        // ---- nonlinear phase modulation ----
        phase_kernel<<<(BATCH * NFFT) / 256, 256, 0, stream>>>(u, scb);
    }

    // ---- final complex conv1d -> d_out [B, 512, 2] complex ----
    conv_kernel<<<(BATCH * 512) / 256, 256, 0, stream>>>(
        u, Cwr, Cwi, Cbr, Cbi, (float2*)d_out);
}
